// TPALSTM_17205638987874
// MI455X (gfx1250) — matrix operand order repaired and checked by reading, not compiled, not run on hardware
//
#include <hip/hip_runtime.h>
#include <stdint.h>

// Problem constants (match reference)
#define B_  128
#define C_  8
#define N_  256
#define T_  64
#define HD  256
#define F_  32

typedef __attribute__((ext_vector_type(16))) __bf16 v16bf;
typedef __attribute__((ext_vector_type(8)))  float  v8f;
typedef __attribute__((ext_vector_type(4)))  int    v4i_;

// gfx1250 async global->LDS staging (guarded; falls back to sync staging)
#if defined(__has_builtin)
#if __has_builtin(__builtin_amdgcn_global_load_async_to_lds_b128) && \
    __has_builtin(__builtin_amdgcn_s_wait_asynccnt)
#define HAVE_ASYNC_LDS 1
#endif
#endif
#ifndef HAVE_ASYNC_LDS
#define HAVE_ASYNC_LDS 0
#endif

#if HAVE_ASYNC_LDS
typedef __attribute__((address_space(1))) v4i_* gptr_t;   // global int4*
typedef __attribute__((address_space(3))) v4i_* lptr_t;   // LDS int4*
#endif

__device__ __forceinline__ uint16_t f2bf(float f) {
  uint32_t u = __float_as_uint(f);
  uint32_t r = u + 0x7fffu + ((u >> 16) & 1u);   // round-to-nearest-even
  return (uint16_t)(r >> 16);
}
__device__ __forceinline__ float sigm_(float x) { return 1.0f / (1.0f + __expf(-x)); }

// ---------------------------------------------------------------------------
// Weight prep: pack W^T (Bt[k][n] = W[n][k]) into per-lane WMMA B fragments.
// dword index (((s*numNt+nt)*32+lane)*8+j); n = nt*16+lane%16;
// k0 = s*32 + (lane/16)*16 + j*2 (contiguous k order, same as the A fragment).
// ---------------------------------------------------------------------------
__global__ void k_swizzleW(uint32_t* __restrict__ dst,
                           const float* __restrict__ srcA, int ldA,
                           const float* __restrict__ srcB, int ldB,
                           int Ksplit, int numNt, int total) {
  int idx = blockIdx.x * blockDim.x + threadIdx.x;
  if (idx >= total) return;
  int s  = idx / (numNt * 256);
  int r  = idx % (numNt * 256);
  int nt = r / 256;
  int r2 = r % 256;
  int l  = r2 >> 3;
  int j  = r2 & 7;
  int n    = nt * 16 + (l & 15);
  int half = l >> 4;
  int k0   = s * 32 + (half << 4) + (j << 1);
  float w0 = (k0     < Ksplit) ? srcA[(size_t)n * ldA + k0]
                               : srcB[(size_t)n * ldB + (k0     - Ksplit)];
  float w1 = (k0 + 1 < Ksplit) ? srcA[(size_t)n * ldA + (k0 + 1)]
                               : srcB[(size_t)n * ldB + (k0 + 1 - Ksplit)];
  dst[idx] = (uint32_t)f2bf(w0) | ((uint32_t)f2bf(w1) << 16);
}

// All 64 rotations of the attention conv kernel, pre-swizzled (256 KB total).
// Circular Hbuf slot p holds logical t = (p - wcnt) & 63.
__global__ void k_rotwc_all(uint32_t* __restrict__ dst, const float* __restrict__ Wc) {
  int idx = blockIdx.x * blockDim.x + threadIdx.x;
  if (idx >= 64 * 1024) return;
  int wcnt = idx >> 10;
  int r0 = idx & 1023;
  int s  = r0 >> 9;
  int r  = r0 & 511;
  int nt = r >> 8;
  int r2 = r & 255;
  int l  = r2 >> 3;
  int j  = r2 & 7;
  int f    = nt * 16 + (l & 15);
  int half = l >> 4;
  int k0   = s * 32 + (half << 4) + (j << 1);
  float w0 = Wc[f * T_ + ((k0     - wcnt) & 63)];
  float w1 = Wc[f * T_ + ((k0 + 1 - wcnt) & 63)];
  dst[idx] = (uint32_t)f2bf(w0) | ((uint32_t)f2bf(w1) << 16);
}

__global__ void k_addbias(float* __restrict__ bg, const float* __restrict__ bih,
                          const float* __restrict__ bhh) {
  int i = blockIdx.x * blockDim.x + threadIdx.x;
  if (i < 4 * HD) bg[i] = bih[i] + bhh[i];
}

// st_conv 1x1 + relu, stored as bf16 A-matrix rows (b*T+t, n)
__global__ void k_conv(const float* __restrict__ x, const float* __restrict__ wst,
                       const float* __restrict__ bst, uint16_t* __restrict__ ybf) {
  int idx = blockIdx.x * blockDim.x + threadIdx.x;   // (b, n, t), t fastest
  if (idx >= B_ * N_ * T_) return;
  int t = idx & 63;
  int n = (idx >> 6) & 255;
  int b = idx >> 14;
  float s = bst[0];
#pragma unroll
  for (int c = 0; c < C_; ++c)
    s += x[(((size_t)(b * C_ + c)) * N_ + n) * T_ + t] * wst[c];
  ybf[((size_t)(b * T_ + t)) * N_ + n] = f2bf(fmaxf(s, 0.f));
}

__global__ void k_lstm_cell(const float* __restrict__ g, float* __restrict__ c,
                            uint16_t* __restrict__ h_bf, uint16_t* __restrict__ htt2,
                            uint16_t* __restrict__ concat2) {
  int i = blockIdx.x * blockDim.x + threadIdx.x;
  if (i >= B_ * HD) return;
  int b = i >> 8, hcol = i & 255;
  const float* gb = g + (size_t)b * 4 * HD;
  float gi = gb[hcol], gf = gb[HD + hcol], gg = gb[2 * HD + hcol], go = gb[3 * HD + hcol];
  float c2 = sigm_(gf) * c[i] + sigm_(gi) * tanhf(gg);
  float h2 = sigm_(go) * tanhf(c2);
  c[i] = c2;
  h_bf[i] = f2bf(h2);
  htt2[(size_t)b * 2 * HD + hcol]      = f2bf(c2);
  htt2[(size_t)b * 2 * HD + HD + hcol] = f2bf(h2);
  concat2[(size_t)b * (2 * HD + F_) + hcol]      = f2bf(fmaxf(c2, 0.f));
  concat2[(size_t)b * (2 * HD + F_) + HD + hcol] = f2bf(fmaxf(h2, 0.f));
}

// alpha = sigmoid(sum_f cv*w); v = sum_o alpha*cv; concat2[.,512:544] = relu(v)
__global__ __launch_bounds__(256) void k_attn_reduce(const float* __restrict__ cv,
                                                     const float* __restrict__ w,
                                                     uint16_t* __restrict__ concat2) {
  __shared__ float sv[F_];
  int b = blockIdx.x, o = threadIdx.x;
  if (o < F_) sv[o] = 0.f;
  __syncthreads();
  const float* cvb = cv + (size_t)b * HD * F_ + (size_t)o * F_;
  const float* wb  = w + (size_t)b * F_;
  float cl[F_];
  float s = 0.f;
#pragma unroll
  for (int f = 0; f < F_; ++f) { cl[f] = cvb[f]; s += cl[f] * wb[f]; }
  float alpha = sigm_(s);
#pragma unroll
  for (int f = 0; f < F_; ++f) atomicAdd(&sv[f], alpha * cl[f]);   // ds_add_f32
  __syncthreads();
  if (o < F_)
    concat2[(size_t)b * (2 * HD + F_) + 2 * HD + o] = f2bf(fmaxf(sv[o], 0.f));
}

__global__ void k_final(const float* __restrict__ res, const float* __restrict__ wend,
                        const float* __restrict__ bend, float* __restrict__ out) {
  size_t idx = (size_t)blockIdx.x * blockDim.x + threadIdx.x;
  if (idx >= (size_t)B_ * C_ * HD * T_) return;
  int t  = idx & 63;
  int h  = (idx >> 6) & 255;
  int ch = (idx >> 14) & 7;
  int b  = (int)(idx >> 17);
  float r = fmaxf(res[((size_t)t * B_ + b) * HD + h], 0.f);
  out[idx] = r * wend[ch] + bend[ch];
}

// ---------------------------------------------------------------------------
// WMMA GEMM with fused concat (dual-source A, split at Ksplit) and LDS-staged
// B panels (shared by all 4 waves of a block; async double-buffered).
// Block = 4 waves, all with the same N-group; wave w handles M-tile mtg*4+w.
// Requires (M/16) % 4 == 0 so no wave early-exits (barrier + WMMA EXEC safe).
// ---------------------------------------------------------------------------
template <int NT, int ACT, bool BIAS, bool WF32, bool WBF16, bool WHBT>
__global__ __launch_bounds__(128) void k_gemm(
    const uint16_t* __restrict__ A0, int lda0, long long bsA,
    const uint16_t* __restrict__ A1, int lda1, int Ksplit,
    const uint32_t* __restrict__ Bsw,
    const float* __restrict__ bias,
    float* __restrict__ Cf, int ldc, long long bsC,
    uint16_t* __restrict__ Cb, int ldcb,
    uint16_t* __restrict__ Hbt, int slot,
    int M, int N, int K) {
  constexpr int NCH = (NT * 64) / 128;   // 16B async chunks per thread per buffer
  static_assert((NT * 64) % 128 == 0, "chunking");
  __shared__ __align__(16) uint32_t Bs[2][NT * 256];

  const int ngroups = N / (16 * NT);
  const int numNt   = N / 16;
  const int Ks      = K / 32;
  int warp = threadIdx.x >> 5;
  int lane = threadIdx.x & 31;
  int ng   = blockIdx.x % ngroups;
  int mt   = (blockIdx.x / ngroups) * 4 + warp;
  int bidx = blockIdx.y;
  const uint16_t* A0row = A0 + (size_t)bidx * bsA + (size_t)((mt << 4) + (lane & 15)) * lda0;
  const uint16_t* A1row = A1 + (size_t)((mt << 4) + (lane & 15)) * lda1;
  int half = lane >> 4;

  auto stageB = [&](int s, int buf) {
    const char* gbase = (const char*)(Bsw + (((size_t)s * numNt + (size_t)ng * NT) << 8));
#pragma unroll
    for (int i = 0; i < NCH; ++i) {
      int c = threadIdx.x + (i << 7);
#if HAVE_ASYNC_LDS
      __builtin_amdgcn_global_load_async_to_lds_b128(
          (gptr_t)(gbase + c * 16), (lptr_t)((char*)&Bs[buf][0] + c * 16), 0, 0);
#else
      *(uint4*)((char*)&Bs[buf][0] + c * 16) = *(const uint4*)(gbase + c * 16);
#endif
    }
  };

  v8f acc[NT];
  v8f zero = {0.f, 0.f, 0.f, 0.f, 0.f, 0.f, 0.f, 0.f};
#pragma unroll
  for (int q = 0; q < NT; ++q) acc[q] = zero;

  stageB(0, 0);
  for (int s = 0; s < Ks; ++s) {
    int buf = s & 1;
    if (s + 1 < Ks) stageB(s + 1, buf ^ 1);

    // A fragment (contiguous k = 16*half + element, same as B); whole 32-K panel comes from one source
    union { uint32_t u[8]; v16bf v; } a;
    int kb = s << 5;
    const uint16_t* base = (kb < Ksplit) ? (A0row + kb) : (A1row + (kb - Ksplit));
#pragma unroll
    for (int j = 0; j < 8; ++j) {
      int k0 = (half << 4) + (j << 1);
      a.u[j] = *(const uint32_t*)(base + k0);
    }

#if HAVE_ASYNC_LDS
    if (s + 1 < Ks) __builtin_amdgcn_s_wait_asynccnt(NCH);  // drain current buf
    else            __builtin_amdgcn_s_wait_asynccnt(0);
#endif
    __syncthreads();                                        // buf ready for all waves
#pragma unroll
    for (int q = 0; q < NT; ++q) {
      const uint32_t* bp = &Bs[buf][q * 256 + lane * 8];
      union { uint32_t u[8]; v16bf v; } bb;
      uint4 lo = *(const uint4*)bp;          // ds_load_b128
      uint4 hi = *(const uint4*)(bp + 4);
      bb.u[0] = lo.x; bb.u[1] = lo.y; bb.u[2] = lo.z; bb.u[3] = lo.w;
      bb.u[4] = hi.x; bb.u[5] = hi.y; bb.u[6] = hi.z; bb.u[7] = hi.w;
      acc[q] = __builtin_amdgcn_wmma_f32_16x16x32_bf16(
          false, a.v, false, bb.v, (short)0, acc[q], false, false);
    }
    __syncthreads();                         // everyone done reading buf
  }

  // C/D layout: VGPR r -> M = half*8 + r, N = lane%16
#pragma unroll
  for (int q = 0; q < NT; ++q) {
    int nbase = (ng * NT + q) << 4;
#pragma unroll
    for (int r = 0; r < 8; ++r) {
      int m = (mt << 4) + (half << 3) + r;
      int n = nbase + (lane & 15);
      float val = acc[q][r];
      if (BIAS) val += bias[n];
      float av = (ACT == 1) ? fmaxf(val, 0.f) : val;
      if (WF32)  Cf[(size_t)bidx * bsC + (size_t)m * ldc + n] = av;
      if (WBF16) Cb[(size_t)m * ldcb + n] = f2bf(av);
      if (WHBT)  Hbt[((size_t)m * HD + n) * T_ + slot] = f2bf(fmaxf(val, 0.f));
    }
  }
}

static inline unsigned gemm_blocks(int M, int N, int NT) {
  return (unsigned)((N / (16 * NT)) * ((M / 16) / 4));
}

// ---------------------------------------------------------------------------
extern "C" void kernel_launch(void* const* d_in, const int* in_sizes, int n_in,
                              void* d_out, int out_size, void* d_ws, size_t ws_size,
                              hipStream_t stream) {
  (void)in_sizes; (void)n_in; (void)out_size; (void)ws_size;
  const float* x    = (const float*)d_in[0];
  const float* wst  = (const float*)d_in[1];
  const float* bst  = (const float*)d_in[2];
  const float* Wh   = (const float*)d_in[3];
  const float* bh   = (const float*)d_in[4];
  const float* Wih  = (const float*)d_in[5];
  const float* Whh  = (const float*)d_in[6];
  const float* bih  = (const float*)d_in[7];
  const float* bhh  = (const float*)d_in[8];
  const float* Wd   = (const float*)d_in[9];
  const float* bd   = (const float*)d_in[10];
  const float* Wc   = (const float*)d_in[11];
  const float* bc   = (const float*)d_in[12];
  const float* Wl1  = (const float*)d_in[13];
  const float* bl1  = (const float*)d_in[14];
  const float* Wl2  = (const float*)d_in[15];
  const float* bl2  = (const float*)d_in[16];
  const float* wend = (const float*)d_in[17];
  const float* bend = (const float*)d_in[18];
  float* out = (float*)d_out;

  char* base = (char*)d_ws;
  size_t off = 0;
  auto alloc = [&](size_t bytes) -> void* {
    off = (off + 255) & ~(size_t)255;
    void* p = base + off;
    off += bytes;
    return p;
  };
  uint16_t* ybf     = (uint16_t*)alloc((size_t)B_ * T_ * N_ * 2);
  uint16_t* xcon_bf = (uint16_t*)alloc((size_t)B_ * T_ * HD * 2);
  uint32_t* WhT_sw  = (uint32_t*)alloc((size_t)8  * 16 * 256 * 4);
  uint32_t* Wg_sw   = (uint32_t*)alloc((size_t)16 * 64 * 256 * 4);
  uint32_t* WdT_sw  = (uint32_t*)alloc((size_t)16 * 16 * 256 * 4);
  uint32_t* Wl1T_sw = (uint32_t*)alloc((size_t)16 * 2  * 256 * 4);
  uint32_t* Wl2T_sw = (uint32_t*)alloc((size_t)17 * 16 * 256 * 4);
  uint32_t* Wc_all  = (uint32_t*)alloc((size_t)64 * 1024 * 4);
  float*    bg      = (float*)alloc((size_t)4 * HD * 4);
  uint16_t* xd_bf   = (uint16_t*)alloc((size_t)B_ * HD * 2);
  float*    gbuf    = (float*)alloc((size_t)B_ * 4 * HD * 4);
  float*    cbuf    = (float*)alloc((size_t)B_ * HD * 4);
  uint16_t* h_bf    = (uint16_t*)alloc((size_t)B_ * HD * 2);
  uint16_t* htt2    = (uint16_t*)alloc((size_t)B_ * 2 * HD * 2);
  float*    wbuf    = (float*)alloc((size_t)B_ * F_ * 4);
  float*    cv      = (float*)alloc((size_t)B_ * HD * F_ * 4);
  uint16_t* concat2 = (uint16_t*)alloc((size_t)B_ * (2 * HD + F_) * 2);
  float*    nh_scr  = (float*)alloc((size_t)B_ * HD * 4);
  uint16_t* attn_bf = (uint16_t*)alloc((size_t)B_ * HD * 2);
  uint16_t* xi_bf   = (uint16_t*)alloc((size_t)B_ * HD * 2);
  uint16_t* HbT     = (uint16_t*)alloc((size_t)B_ * HD * T_ * 2);
  float*    res     = (float*)alloc((size_t)T_ * B_ * HD * 4);

  // deterministic per-call init
  (void)hipMemsetAsync(HbT, 0, (size_t)B_ * HD * T_ * 2, stream);
  (void)hipMemsetAsync(cbuf, 0, (size_t)B_ * HD * 4, stream);
  (void)hipMemsetAsync(h_bf, 0, (size_t)B_ * HD * 2, stream);
  (void)hipMemsetAsync(attn_bf, 0, (size_t)B_ * HD * 2, stream);

  // ---- one-time weight prep ----
  k_swizzleW<<<(8 * 16 * 256 + 255) / 256, 256, 0, stream>>>(WhT_sw, Wh, N_, Wh, N_, 1 << 30, 16, 8 * 16 * 256);
  k_swizzleW<<<(16 * 64 * 256 + 255) / 256, 256, 0, stream>>>(Wg_sw, Wih, HD, Whh, HD, HD, 64, 16 * 64 * 256);
  k_swizzleW<<<(16 * 16 * 256 + 255) / 256, 256, 0, stream>>>(WdT_sw, Wd, 2 * HD, Wd, 2 * HD, 1 << 30, 16, 16 * 16 * 256);
  k_swizzleW<<<(16 * 2 * 256 + 255) / 256, 256, 0, stream>>>(Wl1T_sw, Wl1, 2 * HD, Wl1, 2 * HD, 1 << 30, 2, 16 * 2 * 256);
  k_swizzleW<<<(17 * 16 * 256 + 255) / 256, 256, 0, stream>>>(Wl2T_sw, Wl2, 2 * HD + F_, Wl2, 2 * HD + F_, 1 << 30, 16, 17 * 16 * 256);
  k_rotwc_all<<<(64 * 1024 + 255) / 256, 256, 0, stream>>>(Wc_all, Wc);
  k_addbias<<<4, 256, 0, stream>>>(bg, bih, bhh);

  // ---- phase 0: conv + xcon GEMM (M=8192, K=256, N=256) ----
  k_conv<<<(B_ * N_ * T_ + 255) / 256, 256, 0, stream>>>(x, wst, bst, ybf);
  k_gemm<4, 1, true, false, true, false><<<gemm_blocks(8192, 256, 4), 128, 0, stream>>>(
      ybf, N_, 0, ybf, N_, 1 << 30, WhT_sw, bh,
      nullptr, 0, 0, xcon_bf, HD, nullptr, 0, 8192, 256, 256);

  // ---- sequential scan: T=64 steps x L=2 layers, 8 launches per layer-step --
  for (int t = 0; t < T_; ++t) {
    for (int l = 0; l < 2; ++l) {
      int wcnt = t * 2 + l;
      const uint16_t* xi_src = (l == 0) ? (xcon_bf + (size_t)t * HD) : xi_bf;
      int xi_stride          = (l == 0) ? T_ * HD : HD;

      // xd = [xi, attn] @ Wd^T + bd       (concat fused via dual-source A)
      k_gemm<4, 0, true, false, true, false><<<gemm_blocks(128, 256, 4), 128, 0, stream>>>(
          xi_src, xi_stride, 0, attn_bf, HD, 256, WdT_sw, bd,
          nullptr, 0, 0, xd_bf, HD, nullptr, 0, 128, 256, 512);

      // gates = [xd, h] @ [Wih;Whh]^T + (bih+bhh)
      k_gemm<4, 0, true, true, false, false><<<gemm_blocks(128, 1024, 4), 128, 0, stream>>>(
          xd_bf, HD, 0, h_bf, HD, 256, Wg_sw, bg,
          gbuf, 4 * HD, 0, nullptr, 0, nullptr, 0, 128, 1024, 512);

      k_lstm_cell<<<(B_ * HD + 255) / 256, 256, 0, stream>>>(gbuf, cbuf, h_bf, htt2, concat2);

      // w = [c,h] @ Wl1^T + bl1
      k_gemm<2, 0, true, true, false, false><<<gemm_blocks(128, 32, 2), 128, 0, stream>>>(
          htt2, 2 * HD, 0, htt2, 2 * HD, 1 << 30, Wl1T_sw, bl1,
          wbuf, F_, 0, nullptr, 0, nullptr, 0, 128, 32, 512);

      // cv = relu(Hbuf convolved with rotated Wc + bc), batched over 128
      k_gemm<2, 1, true, true, false, false><<<dim3(gemm_blocks(256, 32, 2), B_), 128, 0, stream>>>(
          HbT, T_, (long long)HD * T_, HbT, T_, 1 << 30,
          Wc_all + (size_t)(wcnt & 63) * 1024, bc,
          cv, F_, (long long)HD * F_, nullptr, 0, nullptr, 0, 256, 32, 64);

      k_attn_reduce<<<B_, 256, 0, stream>>>(cv, wbuf, concat2);

      // new_ht = relu([c,h,v]) @ Wl2^T + bl2  (f32 -> res on last layer)
      float* nh = (l == 1) ? (res + (size_t)t * B_ * HD) : nh_scr;
      k_gemm<4, 0, true, true, true, false><<<gemm_blocks(128, 256, 4), 128, 0, stream>>>(
          concat2, 2 * HD + F_, 0, concat2, 2 * HD + F_, 1 << 30, Wl2T_sw, bl2,
          nh, HD, 0, attn_bf, HD, nullptr, 0, 128, 256, 544);

      // output = [h, new_ht] @ Wd^T + bd ; xi <- output ; HbT[slot] <- relu(output)
      k_gemm<4, 0, true, false, true, true><<<gemm_blocks(128, 256, 4), 128, 0, stream>>>(
          h_bf, HD, 0, attn_bf, HD, 256, WdT_sw, bd,
          nullptr, 0, 0, xi_bf, HD, HbT, wcnt & 63, 128, 256, 512);
    }
  }

  k_final<<<(unsigned)(((size_t)B_ * C_ * HD * T_ + 255) / 256), 256, 0, stream>>>(res, wend, bend, out);
}
